// GNNModel_44813688766875
// MI455X (gfx1250) — matrix operand order repaired and checked by reading, not compiled, not run on hardware
//
#include <hip/hip_runtime.h>

typedef __attribute__((ext_vector_type(16))) _Float16 v16h;
typedef __attribute__((ext_vector_type(8)))  float    v8f;

#define ACT_NONE     0
#define ACT_SOFTPLUS 1
#define ACT_SIGMOID  2

__device__ __forceinline__ float act_apply(float v, int act) {
    if (act == ACT_SOFTPLUS) return fmaxf(v, 0.0f) + log1pf(expf(-fabsf(v)));
    if (act == ACT_SIGMOID)  return 1.0f / (1.0f + expf(-v));
    return v;
}

// ---------------------------------------------------------------------------
// Weight pre-pack: W[K x Nc] fp32 -> f16 B-fragments in CDNA5 lane layout,
// zero-padded out to NTpad n-tiles (multiple of 16) and KT k-tiles, so the
// GEMM needs no tile-activity predicates at all.
// Fragment: elem i -> (K = kt*32 + i + 8g + (i>=8?8:0), N = nt*16 + (lane&15))
// Layout: Bp[((nt*KT + kt)*32 + lane) * 16 halves]
// ---------------------------------------------------------------------------
__global__ void pack_w_kernel(const float* __restrict__ W, _Float16* __restrict__ Bp,
                              int K, int Nc, int KT)
{
    const int kt = blockIdx.x % KT;
    const int nt = blockIdx.x / KT;
    const int lane = threadIdx.x & 31;
    const int g = lane >> 4, ml = lane & 15;
    const int cc = nt * 16 + ml;
    const int ccs = cc < Nc ? cc : Nc - 1;
    v16h frag;
#pragma unroll
    for (int i = 0; i < 16; ++i) {
        int kk = kt * 32 + g * 8 + (i < 8 ? i : i + 8);
        int kks = kk < K ? kk : K - 1;
        float v = W[(size_t)kks * Nc + ccs];
        frag[i] = (kk < K && cc < Nc) ? (_Float16)v : (_Float16)0.0f;
    }
    ((v16h*)Bp)[(size_t)blockIdx.x * 32 + lane] = frag;
}

// ---------------------------------------------------------------------------
// WMMA GEMM: C = act(A * W + bias). A fp32 row-major; W pre-packed/padded f16.
// 4 waves/block, 16 N-tiles/block; wave w owns tiles {w, w+4, w+8, w+12} of
// its block -- unconditionally (padding guarantees fragments exist), so the
// inner loop is branch-free: 1 shared A fragment, 4 in-place WMMAs per K-step.
//   A 16x32 f16 : lane l (g=l>>4, m=l&15), elem i -> (M=m, K = i+8g+(i>=8?8:0))
//   C/D 16x16 f32: elem r -> (M = r+8g, N = l&15)
// ---------------------------------------------------------------------------
__global__ void __launch_bounds__(128)
gemm_wmma_kernel(const float* __restrict__ A, const _Float16* __restrict__ Bp,
                 const float* __restrict__ bias, float* __restrict__ C,
                 int M, int K, int Nc, int KT, int act)
{
    const int wave = threadIdx.x >> 5;
    const int lane = threadIdx.x & 31;
    const int g = lane >> 4, ml = lane & 15;
    const int row0 = blockIdx.x * 16;
    const int rowA = row0 + ml;
    const float* Arow = A + (size_t)(rowA < M ? rowA : M - 1) * K;

    const int nt0 = blockIdx.y * 16 + wave;          // j-th tile = nt0 + 4*j
    const v16h* bp = (const v16h*)Bp + (size_t)nt0 * KT * 32 + lane;
    const size_t jstride = (size_t)4 * KT * 32;      // fragments between j-tiles

    const v8f zero = {0.f, 0.f, 0.f, 0.f, 0.f, 0.f, 0.f, 0.f};
    v8f acc0 = zero, acc1 = zero, acc2 = zero, acc3 = zero;

    const bool a4 = ((K & 3) == 0);
    for (int kt = 0; kt < KT; ++kt) {
        const int k0 = kt * 32;
        v16h a;
        if (a4 && (k0 + 32 <= K)) {                  // fast path: vector loads
            const float4* p0 = (const float4*)(Arow + k0 + 8 * g);
            const float4* p1 = (const float4*)(Arow + k0 + 16 + 8 * g);
            float4 x0 = p0[0], x1 = p0[1], x2 = p1[0], x3 = p1[1];
            if (k0 + 32 < K) __builtin_prefetch(Arow + k0 + 32, 0, 1);
            a[0] = (_Float16)x0.x; a[1] = (_Float16)x0.y;
            a[2] = (_Float16)x0.z; a[3] = (_Float16)x0.w;
            a[4] = (_Float16)x1.x; a[5] = (_Float16)x1.y;
            a[6] = (_Float16)x1.z; a[7] = (_Float16)x1.w;
            a[8] = (_Float16)x2.x; a[9] = (_Float16)x2.y;
            a[10] = (_Float16)x2.z; a[11] = (_Float16)x2.w;
            a[12] = (_Float16)x3.x; a[13] = (_Float16)x3.y;
            a[14] = (_Float16)x3.z; a[15] = (_Float16)x3.w;
        } else {                                     // K-tail: clamp + select
#pragma unroll
            for (int i = 0; i < 16; ++i) {
                int kk = k0 + g * 8 + (i < 8 ? i : i + 8);
                float av = Arow[kk < K ? kk : K - 1];
                a[i] = (kk < K) ? (_Float16)av : (_Float16)0.0f;
            }
        }
        const v16h* bk = bp + (size_t)kt * 32;
        v16h b0 = bk[0];
        v16h b1 = bk[jstride];
        v16h b2 = bk[2 * jstride];
        v16h b3 = bk[3 * jstride];
        acc0 = __builtin_amdgcn_wmma_f32_16x16x32_f16(false, a, false, b0, (short)0, acc0, false, false);
        acc1 = __builtin_amdgcn_wmma_f32_16x16x32_f16(false, a, false, b1, (short)0, acc1, false, false);
        acc2 = __builtin_amdgcn_wmma_f32_16x16x32_f16(false, a, false, b2, (short)0, acc2, false, false);
        acc3 = __builtin_amdgcn_wmma_f32_16x16x32_f16(false, a, false, b3, (short)0, acc3, false, false);
    }

    auto store_tile = [&](const v8f& accv, int nt) {
        int colB = nt * 16 + ml;
        if (colB < Nc) {
            float bv = bias ? bias[colB] : 0.0f;
#pragma unroll
            for (int r = 0; r < 8; ++r) {
                int row = row0 + r + 8 * g;
                if (row < M)
                    C[(size_t)row * Nc + colB] = act_apply(accv[r] + bv, act);
            }
        }
    };
    store_tile(acc0, nt0);
    store_tile(acc1, nt0 + 4);
    store_tile(acc2, nt0 + 8);
    store_tile(acc3, nt0 + 12);
}

// ---------------------------------------------------------------------------
// BatchNorm (training-mode stats, biased variance) + leaky_relu(0.01)
// ---------------------------------------------------------------------------
__global__ void bn_stats_kernel(const float* __restrict__ x, float* __restrict__ stat,
                                int Nn, int C)
{
    __shared__ float ssum[256], ssq[256];
    const int c = blockIdx.x;
    float s = 0.f, q = 0.f;
    for (int n = threadIdx.x; n < Nn; n += blockDim.x) {
        float v = x[(size_t)n * C + c];
        s += v; q += v * v;
    }
    ssum[threadIdx.x] = s; ssq[threadIdx.x] = q;
    __syncthreads();
    for (int off = 128; off > 0; off >>= 1) {
        if ((int)threadIdx.x < off) {
            ssum[threadIdx.x] += ssum[threadIdx.x + off];
            ssq[threadIdx.x]  += ssq[threadIdx.x + off];
        }
        __syncthreads();
    }
    if (threadIdx.x == 0) {
        float mean = ssum[0] / (float)Nn;
        float var  = ssq[0] / (float)Nn - mean * mean;
        stat[c]     = mean;
        stat[C + c] = rsqrtf(var + 1e-5f);
    }
}

__global__ void bn_apply_kernel(const float* __restrict__ x, const float* __restrict__ stat,
                                const float* __restrict__ g, const float* __restrict__ b,
                                float* __restrict__ y, int total, int C, float slope)
{
    int i = blockIdx.x * blockDim.x + threadIdx.x;
    if (i >= total) return;
    int c = i % C;
    float v = (x[i] - stat[c]) * stat[C + c] * g[c] + b[c];
    y[i] = v > 0.f ? v : slope * v;
}

// ---------------------------------------------------------------------------
// Degree (with self-loops) and GCN aggregation
// ---------------------------------------------------------------------------
__global__ void deg_init_kernel(float* deg, int n) {
    int i = blockIdx.x * blockDim.x + threadIdx.x;
    if (i < n) deg[i] = 1.0f;                     // self-loop
}
__global__ void deg_acc_kernel(const int* __restrict__ dst, float* deg, int E_) {
    int i = blockIdx.x * blockDim.x + threadIdx.x;
    if (i < E_) atomicAdd(&deg[dst[i]], 1.0f);
}
__global__ void deg_rsqrt_kernel(float* deg, int n) {
    int i = blockIdx.x * blockDim.x + threadIdx.x;
    if (i < n) deg[i] = rsqrtf(deg[i]);
}
__global__ void gcn_init_kernel(const float* __restrict__ h, const float* __restrict__ dis,
                                const float* __restrict__ bias, float* __restrict__ out,
                                int total, int C)
{
    int i = blockIdx.x * blockDim.x + threadIdx.x;
    if (i >= total) return;
    int n = i / C, c = i % C;
    out[i] = bias[c] + h[i] * dis[n] * dis[n];    // self-loop term
}
__global__ void gcn_scatter_kernel(const float* __restrict__ h, const float* __restrict__ dis,
                                   const int* __restrict__ src, const int* __restrict__ dst,
                                   float* out, int total, int C)
{
    int i = blockIdx.x * blockDim.x + threadIdx.x;
    if (i >= total) return;
    int e = i / C, c = i % C;
    int s = src[e], d = dst[e];
    atomicAdd(&out[(size_t)d * C + c], h[(size_t)s * C + c] * dis[s] * dis[d]);
}

// ---------------------------------------------------------------------------
// GAT: per-dst softmax over edges (incl. self-loops), mean over heads
// ---------------------------------------------------------------------------
__device__ __forceinline__ unsigned fkey(float f) {
    unsigned u = __float_as_uint(f);
    return (u & 0x80000000u) ? ~u : (u | 0x80000000u);
}
__device__ __forceinline__ float funkey(unsigned k) {
    unsigned u = (k & 0x80000000u) ? (k & 0x7fffffffu) : ~k;
    return __uint_as_float(u);
}
__device__ __forceinline__ float lrelu02(float v) { return v > 0.f ? v : 0.2f * v; }

__global__ void gat_node_al_kernel(const float* __restrict__ hfeat,
                                   const float* __restrict__ a_s, const float* __restrict__ a_d,
                                   float* als, float* ald, unsigned* mkey,
                                   int total, int H, int C)
{
    int i = blockIdx.x * blockDim.x + threadIdx.x;
    if (i >= total) return;
    int n = i / H, h = i % H;
    const float* hp = hfeat + ((size_t)n * H + h) * C;
    float ss = 0.f, dd = 0.f;
    for (int c = 0; c < C; ++c) {
        float v = hp[c];
        ss += v * a_s[h * C + c];
        dd += v * a_d[h * C + c];
    }
    als[i] = ss; ald[i] = dd;
    mkey[i] = fkey(lrelu02(ss + dd));             // self-loop seeds the max
}
__global__ void gat_edge_max_kernel(const float* __restrict__ als, const float* __restrict__ ald,
                                    const int* __restrict__ src, const int* __restrict__ dst,
                                    unsigned* mkey, float* ebuf, int total, int H)
{
    int i = blockIdx.x * blockDim.x + threadIdx.x;
    if (i >= total) return;
    int e = i / H, h = i % H;
    int s = src[e], d = dst[e];
    float v = lrelu02(als[s * H + h] + ald[d * H + h]);
    ebuf[i] = v;
    atomicMax(&mkey[d * H + h], fkey(v));
}
__global__ void gat_node_denom_kernel(const float* __restrict__ als, const float* __restrict__ ald,
                                      const unsigned* mkey, float* mfloat,
                                      float* selfex, float* denom, int total)
{
    int i = blockIdx.x * blockDim.x + threadIdx.x;
    if (i >= total) return;
    float m  = funkey(mkey[i]);
    float ex = expf(lrelu02(als[i] + ald[i]) - m);
    mfloat[i] = m;                                // overwrites key buffer (float view)
    selfex[i] = ex;
    denom[i]  = ex;                               // self-loop contribution
}
__global__ void gat_edge_sum_kernel(const float* __restrict__ mfloat,
                                    const int* __restrict__ dst,
                                    float* denom, float* ebuf, int total, int H)
{
    int i = blockIdx.x * blockDim.x + threadIdx.x;
    if (i >= total) return;
    int e = i / H, h = i % H;
    int d = dst[e];
    float ex = expf(ebuf[i] - mfloat[d * H + h]);
    ebuf[i] = ex;
    atomicAdd(&denom[(size_t)d * H + h], ex);
}
__global__ void gat_out_init_kernel(const float* __restrict__ hfeat,
                                    const float* __restrict__ selfex,
                                    const float* __restrict__ denom,
                                    const float* __restrict__ bias,
                                    float* out, int total, int H, int C)
{
    int i = blockIdx.x * blockDim.x + threadIdx.x;
    if (i >= total) return;
    int n = i / C, c = i % C;
    float acc = 0.f;
    for (int h = 0; h < H; ++h)
        acc += selfex[n * H + h] / (denom[n * H + h] + 1e-16f)
             * hfeat[((size_t)n * H + h) * C + c];
    out[i] = bias[c] + acc / (float)H;
}
__global__ void gat_out_scatter_kernel(const float* __restrict__ hfeat,
                                       const float* __restrict__ ebuf,
                                       const float* __restrict__ denom,
                                       const int* __restrict__ src, const int* __restrict__ dst,
                                       float* out, int total, int H, int C)
{
    int i = blockIdx.x * blockDim.x + threadIdx.x;
    if (i >= total) return;
    int e = i / C, c = i % C;
    int s = src[e], d = dst[e];
    float acc = 0.f;
    for (int h = 0; h < H; ++h)
        acc += ebuf[(size_t)e * H + h] / (denom[(size_t)d * H + h] + 1e-16f)
             * hfeat[((size_t)s * H + h) * C + c];
    atomicAdd(&out[(size_t)d * C + c], acc / (float)H);
}

__global__ void concat_kernel(const float* __restrict__ a, const float* __restrict__ b,
                              float* __restrict__ out, int total, int C1, int C2)
{
    int i = blockIdx.x * blockDim.x + threadIdx.x;
    if (i >= total) return;
    int Ct = C1 + C2;
    int n = i / Ct, j = i % Ct;
    out[i] = (j < C1) ? a[(size_t)n * C1 + j] : b[(size_t)n * C2 + (j - C1)];
}

// ---------------------------------------------------------------------------
// Host-side orchestration
// ---------------------------------------------------------------------------
static inline dim3 ewg(long long total) { return dim3((unsigned)((total + 255) / 256)); }

static void launch_gemm(const float* A, const float* W, const float* bias, float* C,
                        int M, int K, int Nc, int act, _Float16* wpack, hipStream_t s)
{
    int KT  = (K + 31) / 32;
    int NTb = (Nc + 255) / 256;          // blocks of 16 n-tiles
    int NTpad = NTb * 16;
    pack_w_kernel<<<dim3(NTpad * KT), dim3(32), 0, s>>>(W, wpack, K, Nc, KT);
    dim3 grid((M + 15) / 16, NTb);
    gemm_wmma_kernel<<<grid, dim3(128), 0, s>>>(A, wpack, bias, C, M, K, Nc, KT, act);
}

static void run_bn_inplace(float* x, const float* g, const float* b, float* stat,
                           int Nn, int C, hipStream_t s)
{
    bn_stats_kernel<<<dim3(C), 256, 0, s>>>(x, stat, Nn, C);
    bn_apply_kernel<<<ewg((long long)Nn * C), 256, 0, s>>>(x, stat, g, b, x, Nn * C, C, 0.01f);
}

static void run_gcn(const float* xin, const float* W, const float* bias, float* out,
                    int Nn, int E_, const int* src, const int* dst, const float* dis,
                    int Kin, int C, float* Hbuf, _Float16* wpack, hipStream_t s)
{
    launch_gemm(xin, W, nullptr, Hbuf, Nn, Kin, C, ACT_NONE, wpack, s);
    gcn_init_kernel<<<ewg((long long)Nn * C), 256, 0, s>>>(Hbuf, dis, bias, out, Nn * C, C);
    gcn_scatter_kernel<<<ewg((long long)E_ * C), 256, 0, s>>>(Hbuf, dis, src, dst, out, E_ * C, C);
}

static void run_gat(const float* xin, const float* W, const float* a_s, const float* a_d,
                    const float* bias, float* out, int Nn, int E_,
                    const int* src, const int* dst, int Kin, int C, int H,
                    float* HFEAT, float* ALS, float* ALD, unsigned* MKEY,
                    float* SELFEX, float* DENOM, float* EBUF, _Float16* wpack, hipStream_t s)
{
    launch_gemm(xin, W, nullptr, HFEAT, Nn, Kin, H * C, ACT_NONE, wpack, s);
    int nh = Nn * H, eh = E_ * H;
    gat_node_al_kernel<<<ewg(nh), 256, 0, s>>>(HFEAT, a_s, a_d, ALS, ALD, MKEY, nh, H, C);
    gat_edge_max_kernel<<<ewg(eh), 256, 0, s>>>(ALS, ALD, src, dst, MKEY, EBUF, eh, H);
    gat_node_denom_kernel<<<ewg(nh), 256, 0, s>>>(ALS, ALD, MKEY, (float*)MKEY, SELFEX, DENOM, nh);
    gat_edge_sum_kernel<<<ewg(eh), 256, 0, s>>>((const float*)MKEY, dst, DENOM, EBUF, eh, H);
    gat_out_init_kernel<<<ewg((long long)Nn * C), 256, 0, s>>>(HFEAT, SELFEX, DENOM, bias, out, Nn * C, H, C);
    gat_out_scatter_kernel<<<ewg((long long)E_ * C), 256, 0, s>>>(HFEAT, EBUF, DENOM, src, dst, out, E_ * C, H, C);
}

extern "C" void kernel_launch(void* const* d_in, const int* in_sizes, int n_in,
                              void* d_out, int out_size, void* d_ws, size_t ws_size,
                              hipStream_t stream)
{
    (void)n_in; (void)out_size; (void)ws_size;
    const int INDIM = 64, OUTDIM = 128, H = 5;
    const int d8 = OUTDIM / 8, d4 = OUTDIM / 4, d2 = OUTDIM / 2, d34 = 3 * OUTDIM / 4;
    const int cat = OUTDIM + d8;                  // 144

    const float* x_in = (const float*)d_in[0];
    const int*   ei   = (const int*)d_in[1];
    const int Nn = in_sizes[0] / INDIM;
    const int E_ = in_sizes[1] / 2;
    const int* src = ei;
    const int* dst = ei + E_;

    const float* P[42];
    for (int i = 2; i < 42; ++i) P[i] = (const float*)d_in[i];
    // nn1: 2..7 | bn0: 8,9 | gcn1: 10,11 | gat1: 12..15 | bn1: 16,17
    // gcn2: 18,19 | gat2: 20..23 | bn2: 24,25 | gcn3: 26,27 | gat3: 28..31
    // bn3: 32,33 | node_pred: 34..41

    // --- workspace layout (floats) ---
    float* ws = (float*)d_ws;
    size_t o = 0;
    float* BUFA = ws + o; o += (size_t)Nn * 128;
    float* BUFB = ws + o; o += (size_t)Nn * 128;
    float* X16  = ws + o; o += (size_t)Nn * d8;
    float* X1   = ws + o; o += (size_t)Nn * d4;
    float* X2   = ws + o; o += (size_t)Nn * d34;
    float* X3   = ws + o; o += (size_t)Nn * OUTDIM;
    float* DIS  = ws + o; o += (size_t)Nn;
    float* STAT = ws + o; o += 256;
    float* ALS  = ws + o; o += (size_t)Nn * H;
    float* ALD  = ws + o; o += (size_t)Nn * H;
    unsigned* MKEY = (unsigned*)(ws + o); o += (size_t)Nn * H;
    float* SELFEX = ws + o; o += (size_t)Nn * H;
    float* DENOM  = ws + o; o += (size_t)Nn * H;
    float* EBUF   = ws + o; o += (size_t)E_ * H;
    _Float16* WPACK = (_Float16*)(ws + o); o += 131072;     // 512 KB f16 pack area (max padded pack ~98K halves)
    float* HFEAT  = ws + o; o += (size_t)Nn * (H * OUTDIM);

    float* out_xf    = (float*)d_out;                       // N x 144
    float* out_probs = (float*)d_out + (size_t)Nn * cat;    // N x 1

    // ---- nn1 MLP: 64 ->sp 128 ->sp 32 -> 16 ----
    launch_gemm(x_in, P[2], P[3], BUFA, Nn, INDIM, 2 * INDIM, ACT_SOFTPLUS, WPACK, stream);
    launch_gemm(BUFA, P[4], P[5], BUFB, Nn, 2 * INDIM, d4, ACT_SOFTPLUS, WPACK, stream);
    launch_gemm(BUFB, P[6], P[7], BUFA, Nn, d4, d8, ACT_NONE, WPACK, stream);
    // bn0 + leaky -> X16
    bn_stats_kernel<<<dim3(d8), 256, 0, stream>>>(BUFA, STAT, Nn, d8);
    bn_apply_kernel<<<ewg((long long)Nn * d8), 256, 0, stream>>>(BUFA, STAT, P[8], P[9], X16,
                                                                 Nn * d8, d8, 0.01f);
    // ---- degree / symmetric normalization (shared by all GCN layers) ----
    deg_init_kernel<<<ewg(Nn), 256, 0, stream>>>(DIS, Nn);
    deg_acc_kernel<<<ewg(E_), 256, 0, stream>>>(dst, DIS, E_);
    deg_rsqrt_kernel<<<ewg(Nn), 256, 0, stream>>>(DIS, Nn);

    // ---- block 1 ----
    run_gcn(X16, P[10], P[11], BUFB, Nn, E_, src, dst, DIS, d8, d4, BUFA, WPACK, stream);
    run_gat(BUFB, P[12], P[13], P[14], P[15], X1, Nn, E_, src, dst, d4, d4, H,
            HFEAT, ALS, ALD, MKEY, SELFEX, DENOM, EBUF, WPACK, stream);
    run_bn_inplace(X1, P[16], P[17], STAT, Nn, d4, stream);

    // ---- block 2 ----
    concat_kernel<<<ewg((long long)Nn * (d8 + d4)), 256, 0, stream>>>(X16, X1, BUFB,
                                                                      Nn * (d8 + d4), d8, d4);
    run_gcn(BUFB, P[18], P[19], BUFB, Nn, E_, src, dst, DIS, d8 + d4, d2, BUFA, WPACK, stream);
    run_gat(BUFB, P[20], P[21], P[22], P[23], X2, Nn, E_, src, dst, d2, d34, H,
            HFEAT, ALS, ALD, MKEY, SELFEX, DENOM, EBUF, WPACK, stream);
    run_bn_inplace(X2, P[24], P[25], STAT, Nn, d34, stream);

    // ---- block 3 ----
    concat_kernel<<<ewg((long long)Nn * (d4 + d34)), 256, 0, stream>>>(X1, X2, BUFB,
                                                                       Nn * (d4 + d34), d4, d34);
    run_gcn(BUFB, P[26], P[27], BUFB, Nn, E_, src, dst, DIS, d4 + d34, OUTDIM, BUFA, WPACK, stream);
    run_gat(BUFB, P[28], P[29], P[30], P[31], X3, Nn, E_, src, dst, OUTDIM, OUTDIM, H,
            HFEAT, ALS, ALD, MKEY, SELFEX, DENOM, EBUF, WPACK, stream);
    run_bn_inplace(X3, P[32], P[33], STAT, Nn, OUTDIM, stream);

    // ---- xf = concat(x, x3) -> d_out ----
    concat_kernel<<<ewg((long long)Nn * cat), 256, 0, stream>>>(X16, X3, out_xf,
                                                                Nn * cat, d8, OUTDIM);
    // ---- node_pred MLP: 144 ->sp 72 ->sp 36 ->sp 18 -> 1, sigmoid ----
    launch_gemm(out_xf, P[34], P[35], BUFA, Nn, cat, cat / 2, ACT_SOFTPLUS, WPACK, stream);
    launch_gemm(BUFA, P[36], P[37], BUFB, Nn, cat / 2, cat / 4, ACT_SOFTPLUS, WPACK, stream);
    launch_gemm(BUFB, P[38], P[39], BUFA, Nn, cat / 4, cat / 8, ACT_SOFTPLUS, WPACK, stream);
    launch_gemm(BUFA, P[40], P[41], out_probs, Nn, cat / 8, 1, ACT_SIGMOID, WPACK, stream);
}
